// UNetBlock_57698590654673
// MI455X (gfx1250) — matrix operand order repaired and checked by reading, not compiled, not run on hardware
//
#include <hip/hip_runtime.h>

#define DIMC   256
#define CTXD   768
#define BATCH  16
#define HSZ    64
#define WSZ    64
#define HW     4096
#define LCTX   77
#define LPAD   96
#define NGROUP 8
#define EPSF   1e-5f

typedef __bf16 bf16_t;
typedef __attribute__((ext_vector_type(16))) __bf16        v16bf;
typedef __attribute__((ext_vector_type(8)))  float         v8f;
typedef __attribute__((ext_vector_type(4)))  unsigned int  v4u;
typedef __attribute__((ext_vector_type(8)))  unsigned int  v8u;

__device__ __forceinline__ v8f wmma_bf16(v16bf a, v16bf b, v8f c) {
  return __builtin_amdgcn_wmma_f32_16x16x32_bf16(false, a, false, b, (short)0, c,
                                                 false, false);
}

__device__ __forceinline__ v16bf merge16(v4u lo, v4u hi) {
  const v8u full = __builtin_shufflevector(lo, hi, 0, 1, 2, 3, 4, 5, 6, 7);
  return __builtin_bit_cast(v16bf, full);
}

// A fragment (16x32 bf16): lane holds row m=lane&15; elems j=0..15 at k0..k0+15
// with k0 = kc*32 + 16*(lane>=16). Two b128 loads.
__device__ __forceinline__ v16bf load_afrag(const bf16_t* __restrict__ row,
                                            int kc, int hv) {
  const int k0 = kc * 32 + (hv << 4);
  const v4u lo = *(const v4u*)(row + k0);
  const v4u hi = *(const v4u*)(row + k0 + 8);
  return merge16(lo, hi);
}

// B fragment (32x16 bf16) from row-major [n][k] bf16 weights: lane holds col
// n=lane&15, 16 contiguous k at k0 = kc*32 + 16*(lane>=16). Two b128 loads.
__device__ __forceinline__ v16bf load_bfrag_rows(const bf16_t* __restrict__ row,
                                                 int kc, int hv) {
  const int k0 = kc * 32 + (hv << 4);
  const v4u lo = *(const v4u*)(row + k0);
  const v4u hi = *(const v4u*)(row + k0 + 8);
  return merge16(lo, hi);
}

// B fragment from a pre-packed register image: 512 bf16 block, lane*32 bytes.
__device__ __forceinline__ v16bf load_bfrag_pk(const bf16_t* __restrict__ blk,
                                               int lane) {
  const v4u lo = *(const v4u*)(blk + lane * 16);
  const v4u hi = *(const v4u*)(blk + lane * 16 + 8);
  return merge16(lo, hi);
}

// ---------------------------------------------------------------------------
// Zero row buffer (768 bf16) used for out-of-bounds row pointer selection.
// ---------------------------------------------------------------------------
__global__ __launch_bounds__(256) void k_zero(bf16_t* __restrict__ z) {
  z[blockIdx.x * 256 + threadIdx.x] = (bf16_t)0.0f;
}

// ---------------------------------------------------------------------------
// x (NCHW f32) -> xt (token-major [B,HW,C] bf16)
// ---------------------------------------------------------------------------
__global__ __launch_bounds__(256) void k_x2bf(const float* __restrict__ x,
                                              bf16_t* __restrict__ xt) {
  const size_t idx = (size_t)blockIdx.x * 256 + threadIdx.x;  // B*C*HW
  const int s = (int)(idx & (HW - 1));
  const int c = (int)((idx >> 12) & (DIMC - 1));
  const int b = (int)(idx >> 20);
  xt[((size_t)b * HW + s) * DIMC + c] = (bf16_t)x[idx];
}

// ---------------------------------------------------------------------------
// Generic f32 -> bf16 copy (weights / context).
// ---------------------------------------------------------------------------
__global__ __launch_bounds__(256) void k_tobf(const float* __restrict__ in,
                                              bf16_t* __restrict__ out, int n) {
  const int i = blockIdx.x * 256 + threadIdx.x;
  if (i < n) out[i] = (bf16_t)in[i];
}

// ---------------------------------------------------------------------------
// Pack conv weights (OIHW f32) into per-(cout-tile, tap, k-chunk) WMMA
// B-fragment register images: 512 bf16 per block.
// ---------------------------------------------------------------------------
__global__ __launch_bounds__(256) void k_wpack(const float* __restrict__ cw,
                                               bf16_t* __restrict__ wpk) {
  const int i = blockIdx.x * 256 + threadIdx.x;  // 589824 = 16*9*8*512
  const int e = i & 511;
  const int f = i >> 9;          // (coutt*9 + tap)*8 + kc
  const int kc = f & 7;
  const int tf = f >> 3;
  const int tap = tf % 9;
  const int coutt = tf / 9;
  const int ln = e >> 4, j = e & 15;
  const int n = ln & 15, hv = ln >> 4;
  const int cin = kc * 32 + hv * 16 + j;
  wpk[i] = (bf16_t)cw[((size_t)(coutt * 16 + n) * DIMC + cin) * 9 + tap];
}

// ---------------------------------------------------------------------------
// Conv3x3 implicit GEMM. Per wave: one full image row (64 spatial) x 16 cout;
// four M-tiles share every B fragment -> per (dy,dx,kc): 1 B load + 4 A loads
// + 4 WMMAs (288 WMMAs/wave). Out-of-bounds rows select a zero-row pointer
// (one 64-bit cndmask) instead of masking fragment values.
// ---------------------------------------------------------------------------
__global__ __launch_bounds__(256) void k_conv(const bf16_t* __restrict__ xt,
                                              const bf16_t* __restrict__ wpk,
                                              const float* __restrict__ cb,
                                              const bf16_t* __restrict__ zrow,
                                              float* __restrict__ hconv) {
  const int lane = threadIdx.x & 31;
  const int hv = lane >> 4, mn = lane & 15;
  const int wid = blockIdx.x * 8 + (threadIdx.x >> 5);
  // B(16) x coutt(16) x hrow(64)
  const int b     = wid >> 10;
  const int rem   = wid & 1023;
  const int coutt = rem >> 6;
  const int hrow  = rem & 63;
  const int cout0 = coutt << 4;

  v8f acc[4] = {{}, {}, {}, {}};
  for (int dy = -1; dy <= 1; ++dy) {
    const int ih = hrow + dy;
    const bool hok = (unsigned)ih < 64u;
    const int ihc = min(max(ih, 0), 63);
    const bf16_t* base = xt + ((size_t)b * HW + ihc * WSZ) * DIMC;
    for (int dx = -1; dx <= 1; ++dx) {
      const bf16_t* rp[4];
      #pragma unroll
      for (int t = 0; t < 4; ++t) {
        const int iw = t * 16 + mn + dx;
        const bool ok = hok && ((unsigned)iw < 64u);
        const int iwc = min(max(iw, 0), 63);
        rp[t] = ok ? (base + (size_t)iwc * DIMC) : zrow;
      }
      const bf16_t* wblk =
          wpk + (size_t)((coutt * 9 + (dy + 1) * 3 + (dx + 1)) * 8) * 512;
      #pragma unroll
      for (int kc = 0; kc < 8; ++kc) {
        const v16bf bfr = load_bfrag_pk(wblk + kc * 512, lane);
        #pragma unroll
        for (int t = 0; t < 4; ++t)
          acc[t] = wmma_bf16(load_afrag(rp[t], kc, hv), bfr, acc[t]);
      }
    }
  }
  const float bias = cb[cout0 + mn];
  float* op = hconv + ((size_t)b * DIMC + cout0 + mn) * HW + (size_t)hrow * WSZ;
  #pragma unroll
  for (int t = 0; t < 4; ++t) {
    #pragma unroll
    for (int r = 0; r < 8; ++r)
      op[t * 16 + r + (hv << 3)] = acc[t][r] + bias;
  }
}

// ---------------------------------------------------------------------------
// GroupNorm statistics (per batch*group; group channels contiguous in NCHW).
// ---------------------------------------------------------------------------
__global__ __launch_bounds__(256) void k_gnstats(const float* __restrict__ hconv,
                                                 float* __restrict__ stats) {
  const int bg = blockIdx.x;
  const float* p = hconv + (size_t)bg * (32 * HW);
  float s1 = 0.f, s2 = 0.f;
  for (int i = threadIdx.x; i < 32 * HW; i += 256) {
    const float v = p[i];
    s1 += v; s2 += v * v;
  }
  __shared__ float r1[256], r2[256];
  r1[threadIdx.x] = s1; r2[threadIdx.x] = s2;
  __syncthreads();
  for (int off = 128; off > 0; off >>= 1) {
    if (threadIdx.x < off) {
      r1[threadIdx.x] += r1[threadIdx.x + off];
      r2[threadIdx.x] += r2[threadIdx.x + off];
    }
    __syncthreads();
  }
  if (threadIdx.x == 0) {
    const float inv = 1.0f / (32.0f * HW);
    const float m = r1[0] * inv;
    const float var = r2[0] * inv - m * m;
    stats[bg * 2]     = m;
    stats[bg * 2 + 1] = rsqrtf(var + EPSF);
  }
}

// ---------------------------------------------------------------------------
// Normalize + affine + SiLU, NCHW f32 -> token-major bf16.
// ---------------------------------------------------------------------------
__global__ __launch_bounds__(256) void k_gnsilu(const float* __restrict__ hconv,
                                                const float* __restrict__ stats,
                                                const float* __restrict__ gnw,
                                                const float* __restrict__ gnb,
                                                bf16_t* __restrict__ hf) {
  const size_t idx = (size_t)blockIdx.x * 256 + threadIdx.x;
  const int s = (int)(idx & (HW - 1));
  const int c = (int)((idx >> 12) & (DIMC - 1));
  const int b = (int)(idx >> 20);
  const int g = c >> 5;
  const float m  = stats[(b * NGROUP + g) * 2];
  const float rs = stats[(b * NGROUP + g) * 2 + 1];
  float v = hconv[idx];
  v = (v - m) * rs * gnw[c] + gnb[c];
  v = v / (1.0f + __expf(-v));
  hf[((size_t)b * HW + s) * DIMC + c] = (bf16_t)v;
}

// ---------------------------------------------------------------------------
// q = hf @ wq^T + bq. Per wave 16x32 (two N-tiles share the A fragment).
// ---------------------------------------------------------------------------
__global__ __launch_bounds__(256) void k_projq(const bf16_t* __restrict__ A,
                                               const bf16_t* __restrict__ Wb,
                                               const float* __restrict__ bias,
                                               bf16_t* __restrict__ out) {
  const int lane = threadIdx.x & 31, hv = lane >> 4, mn = lane & 15;
  const int wid = blockIdx.x * 8 + (threadIdx.x >> 5);  // 4096 mt x 8 npair
  const int n0 = (wid & 7) << 5;
  const int m0 = (wid >> 3) << 4;
  const bf16_t* arow  = A  + (size_t)(m0 + mn) * DIMC;
  const bf16_t* w0row = Wb + (size_t)(n0 + mn) * DIMC;
  const bf16_t* w1row = Wb + (size_t)(n0 + 16 + mn) * DIMC;
  v8f acc0 = {}, acc1 = {};
  #pragma unroll
  for (int kc = 0; kc < 8; ++kc) {
    const v16bf a = load_afrag(arow, kc, hv);
    acc0 = wmma_bf16(a, load_bfrag_rows(w0row, kc, hv), acc0);
    acc1 = wmma_bf16(a, load_bfrag_rows(w1row, kc, hv), acc1);
  }
  const float bn0 = bias[n0 + mn], bn1 = bias[n0 + 16 + mn];
  #pragma unroll
  for (int r = 0; r < 8; ++r) {
    const size_t m = (size_t)(m0 + r + (hv << 3));
    out[m * DIMC + n0 + mn]      = (bf16_t)(acc0[r] + bn0);
    out[m * DIMC + n0 + 16 + mn] = (bf16_t)(acc1[r] + bn1);
  }
}

// ---------------------------------------------------------------------------
// k,v = context @ w^T + b (L padded 77->96). Padded rows read the zero row
// via pointer select; shared A fragment between the k and v accumulators.
// ---------------------------------------------------------------------------
__global__ __launch_bounds__(256) void k_kv(const bf16_t* __restrict__ ctxb,
                                            const bf16_t* __restrict__ wkb,
                                            const float* __restrict__ bk,
                                            const bf16_t* __restrict__ wvb,
                                            const float* __restrict__ bv,
                                            const bf16_t* __restrict__ zrow,
                                            bf16_t* __restrict__ kbuf,
                                            bf16_t* __restrict__ vbuf) {
  const int lane = threadIdx.x & 31, hv = lane >> 4, mn = lane & 15;
  const int wid = blockIdx.x * 8 + (threadIdx.x >> 5);  // B(16)*LT(6)*NT(16)
  const int n0 = (wid & 15) << 4;
  const int l0 = ((wid >> 4) % 6) << 4;
  const int b  = wid / 96;
  const int lrow = l0 + mn;
  const bool lok = lrow < LCTX;
  const bf16_t* crow  = lok ? (ctxb + ((size_t)b * LCTX + lrow) * CTXD) : zrow;
  const bf16_t* wkrow = wkb + (size_t)(n0 + mn) * CTXD;
  const bf16_t* wvrow = wvb + (size_t)(n0 + mn) * CTXD;
  v8f acck = {}, accv = {};
  #pragma unroll 4
  for (int kc = 0; kc < 24; ++kc) {
    const v16bf a = load_afrag(crow, kc, hv);
    acck = wmma_bf16(a, load_bfrag_rows(wkrow, kc, hv), acck);
    accv = wmma_bf16(a, load_bfrag_rows(wvrow, kc, hv), accv);
  }
  const float bkn = bk[n0 + mn], bvn = bv[n0 + mn];
  #pragma unroll
  for (int r = 0; r < 8; ++r) {
    const int l = l0 + r + (hv << 3);
    const size_t o = ((size_t)b * LPAD + l) * DIMC + n0 + mn;
    kbuf[o] = (bf16_t)(l < LCTX ? acck[r] + bkn : 0.0f);
    vbuf[o] = (bf16_t)(l < LCTX ? accv[r] + bvn : 0.0f);
  }
}

// ---------------------------------------------------------------------------
// Attention. Per wave one 16-row Q tile; all 8 waves of a block share batch b,
// so V is cooperatively staged per N-tile in LDS pre-swizzled into B-fragment
// layout (ds_load_b128 feeds). S=QK^T 48 WMMAs, masked softmax via shfl_xor
// within 16-lane halves, P restriped through LDS, O=PV 48 WMMAs.
// ---------------------------------------------------------------------------
__global__ __launch_bounds__(256) void k_attn(const bf16_t* __restrict__ q,
                                              const bf16_t* __restrict__ kbuf,
                                              const bf16_t* __restrict__ vbuf,
                                              bf16_t* __restrict__ out) {
  const int lane = threadIdx.x & 31, hv = lane >> 4, mn = lane & 15;
  const int wv_ = threadIdx.x >> 5;
  const int wid = blockIdx.x * 8 + wv_;
  const int b  = wid >> 8;          // 32 blocks per batch -> uniform in block
  const int m0 = (wid & 255) << 4;

  __builtin_prefetch(vbuf + (size_t)b * LPAD * DIMC, 0, 1);

  const bf16_t* qrow = q + ((size_t)b * HW + m0 + mn) * DIMC;
  v16bf qa[8];
  #pragma unroll
  for (int kc = 0; kc < 8; ++kc) qa[kc] = load_afrag(qrow, kc, hv);

  v8f s[6];
  #pragma unroll
  for (int t = 0; t < 6; ++t) {
    v8f acc = {};
    const bf16_t* krow = kbuf + ((size_t)b * LPAD + t * 16 + mn) * DIMC;
    #pragma unroll
    for (int kc = 0; kc < 8; ++kc)
      acc = wmma_bf16(qa[kc], load_bfrag_rows(krow, kc, hv), acc);
    s[t] = acc;
  }

  __shared__ __align__(16) bf16_t ldsP[8][16][104];  // rows 16B-aligned
  __shared__ __align__(16) bf16_t ldsV[3 * 512];     // B-frag images per kc
  const float scale = 0.0625f;  // 1/sqrt(256)
  #pragma unroll
  for (int r = 0; r < 8; ++r) {
    float vals[6];
    float rowm = -3.0e38f;
    #pragma unroll
    for (int t = 0; t < 6; ++t) {
      const int col = t * 16 + mn;
      const float vv = (col < LCTX) ? s[t][r] * scale : -3.0e38f;
      vals[t] = vv;
      rowm = fmaxf(rowm, vv);
    }
    #pragma unroll
    for (int off = 1; off < 16; off <<= 1)
      rowm = fmaxf(rowm, __shfl_xor(rowm, off, 32));
    float rsum = 0.f;
    #pragma unroll
    for (int t = 0; t < 6; ++t) { vals[t] = __expf(vals[t] - rowm); rsum += vals[t]; }
    #pragma unroll
    for (int off = 1; off < 16; off <<= 1)
      rsum += __shfl_xor(rsum, off, 32);
    const float inv = 1.0f / rsum;
    #pragma unroll
    for (int t = 0; t < 6; ++t)
      ldsP[wv_][r + (hv << 3)][t * 16 + mn] = (bf16_t)(vals[t] * inv);
  }
  __syncthreads();

  v16bf pa[3];
  #pragma unroll
  for (int kc = 0; kc < 3; ++kc)
    pa[kc] = load_afrag(&ldsP[wv_][mn][0], kc, hv);

  for (int nt = 0; nt < 16; ++nt) {
    const int n0 = nt << 4;
    __syncthreads();  // previous tile's ldsV reads complete
    // stage V[:, n0:n0+16] as 3 B-fragment register images (1536 elems).
    #pragma unroll
    for (int i = 0; i < 6; ++i) {
      const int e = threadIdx.x + 256 * i;
      const int kc = e >> 9, e2 = e & 511;
      const int ln = e2 >> 4, j = e2 & 15;
      const int k = kc * 32 + (ln >> 4) * 16 + j;
      ldsV[e] = vbuf[((size_t)b * LPAD + k) * DIMC + n0 + (ln & 15)];
    }
    __syncthreads();
    v8f acc = {};
    #pragma unroll
    for (int kc = 0; kc < 3; ++kc)
      acc = wmma_bf16(pa[kc], load_bfrag_pk(ldsV + kc * 512, lane), acc);
    #pragma unroll
    for (int r = 0; r < 8; ++r)
      out[((size_t)b * HW + m0 + r + (hv << 3)) * DIMC + n0 + mn] = (bf16_t)acc[r];
  }
}

// ---------------------------------------------------------------------------
// out = x + (attn @ wo^T + bo), scattered to NCHW f32. 16x32 per wave.
// ---------------------------------------------------------------------------
__global__ __launch_bounds__(256) void k_outproj(const bf16_t* __restrict__ A,
                                                 const bf16_t* __restrict__ wob,
                                                 const float* __restrict__ bo,
                                                 const float* __restrict__ x,
                                                 float* __restrict__ out) {
  const int lane = threadIdx.x & 31, hv = lane >> 4, mn = lane & 15;
  const int wid = blockIdx.x * 8 + (threadIdx.x >> 5);
  const int n0 = (wid & 7) << 5;
  const int m0 = (wid >> 3) << 4;
  const bf16_t* arow  = A   + (size_t)(m0 + mn) * DIMC;
  const bf16_t* w0row = wob + (size_t)(n0 + mn) * DIMC;
  const bf16_t* w1row = wob + (size_t)(n0 + 16 + mn) * DIMC;
  v8f acc0 = {}, acc1 = {};
  #pragma unroll
  for (int kc = 0; kc < 8; ++kc) {
    const v16bf a = load_afrag(arow, kc, hv);
    acc0 = wmma_bf16(a, load_bfrag_rows(w0row, kc, hv), acc0);
    acc1 = wmma_bf16(a, load_bfrag_rows(w1row, kc, hv), acc1);
  }
  const float bn0 = bo[n0 + mn], bn1 = bo[n0 + 16 + mn];
  #pragma unroll
  for (int r = 0; r < 8; ++r) {
    const int m = m0 + r + (hv << 3);
    const int bb = m >> 12, ss = m & (HW - 1);
    const size_t o0 = ((size_t)bb * DIMC + n0 + mn) * HW + ss;
    const size_t o1 = o0 + (size_t)16 * HW;
    out[o0] = x[o0] + acc0[r] + bn0;
    out[o1] = x[o1] + acc1[r] + bn1;
  }
}

extern "C" void kernel_launch(void* const* d_in, const int* in_sizes, int n_in,
                              void* d_out, int out_size, void* d_ws, size_t ws_size,
                              hipStream_t stream) {
  (void)in_sizes; (void)n_in; (void)out_size; (void)ws_size;
  const float* x   = (const float*)d_in[0];
  const float* ctx = (const float*)d_in[1];
  const float* cw  = (const float*)d_in[2];
  const float* cb  = (const float*)d_in[3];
  const float* gnw = (const float*)d_in[4];
  const float* gnb = (const float*)d_in[5];
  const float* wq  = (const float*)d_in[6];
  const float* bq  = (const float*)d_in[7];
  const float* wk  = (const float*)d_in[8];
  const float* bk  = (const float*)d_in[9];
  const float* wv  = (const float*)d_in[10];
  const float* bv  = (const float*)d_in[11];
  const float* wo  = (const float*)d_in[12];
  const float* bo  = (const float*)d_in[13];
  float* out = (float*)d_out;

  char* ws = (char*)d_ws;
  size_t off = 0;
  float*  hconv = (float*)(ws + off);  off += (size_t)BATCH * DIMC * HW * 4;   // 64 MiB
  bf16_t* xt    = (bf16_t*)(ws + off); off += (size_t)BATCH * HW * DIMC * 2;   // 32 MiB
  bf16_t* qb    = (bf16_t*)(ws + off); off += (size_t)BATCH * HW * DIMC * 2;   // 32 MiB
  bf16_t* kb_   = (bf16_t*)(ws + off); off += (size_t)BATCH * LPAD * DIMC * 2;
  bf16_t* vb_   = (bf16_t*)(ws + off); off += (size_t)BATCH * LPAD * DIMC * 2;
  float*  stats = (float*)(ws + off);  off += (size_t)BATCH * NGROUP * 2 * 4;
  bf16_t* wpk   = (bf16_t*)(ws + off); off += (size_t)16 * 9 * 8 * 512 * 2;
  bf16_t* wqb   = (bf16_t*)(ws + off); off += (size_t)DIMC * DIMC * 2;
  bf16_t* wkb   = (bf16_t*)(ws + off); off += (size_t)DIMC * CTXD * 2;
  bf16_t* wvb   = (bf16_t*)(ws + off); off += (size_t)DIMC * CTXD * 2;
  bf16_t* wob   = (bf16_t*)(ws + off); off += (size_t)DIMC * DIMC * 2;
  bf16_t* ctxb  = (bf16_t*)(ws + off); off += (size_t)BATCH * LCTX * CTXD * 2;
  bf16_t* zrow  = (bf16_t*)(ws + off); off += (size_t)CTXD * 2;  // 768 zeros
  bf16_t* hf    = xt;             // xt dead after k_conv -> reuse for hf
  bf16_t* attn  = (bf16_t*)hconv; // hconv dead after k_gnsilu -> reuse

  k_zero <<<3, 256, 0, stream>>>(zrow);
  k_x2bf <<<65536, 256, 0, stream>>>(x, xt);
  k_wpack<<<2304, 256, 0, stream>>>(cw, wpk);
  k_tobf <<<256, 256, 0, stream>>>(wq, wqb, DIMC * DIMC);
  k_tobf <<<768, 256, 0, stream>>>(wk, wkb, DIMC * CTXD);
  k_tobf <<<768, 256, 0, stream>>>(wv, wvb, DIMC * CTXD);
  k_tobf <<<256, 256, 0, stream>>>(wo, wob, DIMC * DIMC);
  k_tobf <<<3696, 256, 0, stream>>>(ctx, ctxb, BATCH * LCTX * CTXD);

  k_conv   <<<2048, 256, 0, stream>>>(xt, wpk, cb, zrow, hconv);
  k_gnstats<<<BATCH * NGROUP, 256, 0, stream>>>(hconv, stats);
  k_gnsilu <<<65536, 256, 0, stream>>>(hconv, stats, gnw, gnb, hf);
  k_projq  <<<4096, 256, 0, stream>>>(hf, wqb, bq, qb);
  k_kv     <<<192, 256, 0, stream>>>(ctxb, wkb, bk, wvb, bv, zrow, kb_, vb_);
  k_attn   <<<512, 256, 0, stream>>>(qb, kb_, vb_, attn);
  k_outproj<<<4096, 256, 0, stream>>>(attn, wob, bo, x, out);
}
